// GeoAttention_33036888441302
// MI455X (gfx1250) — matrix operand order repaired and checked by reading, not compiled, not run on hardware
//
#include <hip/hip_runtime.h>
#include <hip/hip_bf16.h>
#include <stdint.h>

typedef __attribute__((ext_vector_type(16))) _Float16 v16h;
typedef __attribute__((ext_vector_type(8)))  _Float16 h8;
typedef __attribute__((ext_vector_type(8)))  float    v8f;

#define NB      2
#define NSEQ    2048
#define DMODEL  512
#define NHEAD   8
#define DK      64
#define NEG_INF -1e9f

// ---------------------------------------------------------------------------
// Kernel 1: per-head Q/K/V projections, fp32 in -> f16 out (WMMA f16 path).
// One wave computes one 16x16 output tile; K-loop over DMODEL in 32-chunks.
// scale(0.125)*exp(beta[h]) folded into Q. Q/K stored [seq][dk]; V stored
// TRANSPOSED [dk][seq] so the attention P.V B-fragments are contiguous.
// ---------------------------------------------------------------------------
__global__ __launch_bounds__(256) void geo_proj_kernel(
    const float* __restrict__ q, const float* __restrict__ k,
    const float* __restrict__ v,
    const float* __restrict__ q_proj, const float* __restrict__ k_proj,
    const float* __restrict__ v_proj,
    const float* __restrict__ q_bias, const float* __restrict__ k_bias,
    const float* __restrict__ v_bias,
    const float* __restrict__ beta_w,
    _Float16* __restrict__ Qh, _Float16* __restrict__ Kh,
    _Float16* __restrict__ Vt)
{
    int wave = (int)((blockIdx.x * blockDim.x + threadIdx.x) >> 5);
    int lane = (int)(threadIdx.x & 31);
    int half = lane >> 4, lr = lane & 15;

    // decode tile: [which(3)][b(2)][head(8)][rowTile(128)][colTile(4)]
    int t  = wave;
    int ct = t & 3;   t >>= 2;       // DK/16 = 4
    int rt = t & 127; t >>= 7;       // NSEQ/16 = 128
    int hh = t & 7;   t >>= 3;       // heads
    int b  = t & 1;   t >>= 1;
    int which = t;                   // 0=q 1=k 2=v

    const float* X  = (which == 0 ? q : which == 1 ? k : v) + (size_t)b * NSEQ * DMODEL;
    const float* W  = (which == 0 ? q_proj : which == 1 ? k_proj : v_proj) + (size_t)hh * DMODEL * DK;
    const float* bs = (which == 0 ? q_bias : which == 1 ? k_bias : v_bias) + hh * DK;

    float mul = 1.0f;
    if (which == 0) mul = 0.125f * __expf(beta_w[hh]);   // 1/sqrt(DK) * beta

    int rowBase = rt * 16, colBase = ct * 16;
    v8f acc = {};

    for (int kb = 0; kb < DMODEL; kb += 32) {
        // A fragment: row (rowBase+lr), K = kb + half*8 + {0..7, 16..23}
        const float* arow = X + (size_t)(rowBase + lr) * DMODEL + kb + half * 8;
        float4 a0 = *(const float4*)(arow);
        float4 a1 = *(const float4*)(arow + 4);
        float4 a2 = *(const float4*)(arow + 16);
        float4 a3 = *(const float4*)(arow + 20);
        float af[16] = {a0.x,a0.y,a0.z,a0.w, a1.x,a1.y,a1.z,a1.w,
                        a2.x,a2.y,a2.z,a2.w, a3.x,a3.y,a3.z,a3.w};
        v16h A;
#pragma unroll
        for (int i = 0; i < 16; ++i) A[i] = (_Float16)af[i];

        // B fragment: col = colBase+lr, K = kb + half*8 + {0..7, 16..23}  (stride DK)
        const float* wcol = W + (size_t)(kb + half * 8) * DK + colBase + lr;
        v16h Bv;
#pragma unroll
        for (int i = 0; i < 16; ++i) Bv[i] = (_Float16)wcol[(size_t)(i < 8 ? i : i + 8) * DK];

        acc = __builtin_amdgcn_wmma_f32_16x16x32_f16(false, A, false, Bv,
                                                     (short)0, acc, false, false);
    }

    float bv = bs[colBase + lr];
    if (which == 2) {
        // V transposed: Vt[(b,h)][col][seq]; lane's 8 rows are consecutive.
        h8 vec;
#pragma unroll
        for (int r = 0; r < 8; ++r) vec[r] = (_Float16)(acc[r] + bv);
        _Float16* p = Vt + (size_t)(b * NHEAD + hh) * DK * NSEQ
                         + (size_t)(colBase + lr) * NSEQ + rowBase + half * 8;
        *(h8*)p = vec;
    } else {
        _Float16* Out = (which == 0 ? Qh : Kh) + (size_t)(b * NHEAD + hh) * NSEQ * DK;
#pragma unroll
        for (int r = 0; r < 8; ++r) {
            int M = r + half * 8;               // C layout: lanes16-31 -> rows 8..15
            float val = (acc[r] + bv) * mul;
            Out[(size_t)(rowBase + M) * DK + colBase + lr] = (_Float16)val;
        }
    }
}

// ---------------------------------------------------------------------------
// Kernel 2: fused RBF attention, flash-style online softmax.
// All 8 waves of a block share one (b, head): the 32-key K tile is staged in
// a block-shared double buffer in LDS, filled with global_load_async_to_lds
// (ASYNCcnt) by wave 0 and published via __syncthreads(). Per 32-key block:
//   4 WMMA (QK^T, K-dim 64) + RBF/threshold/mask + online softmax
//   + P C->A transpose through per-wave LDS + 4 WMMA (P.V from transposed V).
// ---------------------------------------------------------------------------
__global__ __launch_bounds__(256) void geo_attn_kernel(
    const _Float16* __restrict__ Qh, const _Float16* __restrict__ Kh,
    const _Float16* __restrict__ Vt,
    const float* __restrict__ coords, const unsigned char* __restrict__ mask,
    const float* __restrict__ spread_w,
    _Float16* __restrict__ AO)
{
    __shared__ _Float16 klds[2][32 * DK];    // double-buffered 32-key K tile (8KB)
    __shared__ _Float16 pstage[8][16 * 40];  // per-wave P staging, padded rows

    int wib  = (int)(threadIdx.x >> 5);
    int lane = (int)(threadIdx.x & 31);
    int half = lane >> 4, lr = lane & 15;
    int wave = (int)(blockIdx.x * (blockDim.x >> 5) + wib);

    int qt = wave & 127; wave >>= 7;         // NSEQ/16 query tiles (8 per block)
    int hh = wave & 7;   wave >>= 3;
    int b  = wave & 1;

    size_t bh = (size_t)(b * NHEAD + hh);
    const _Float16* Qp = Qh + bh * NSEQ * DK;
    const _Float16* Kp = Kh + bh * NSEQ * DK;
    const _Float16* Vp = Vt + bh * DK * NSEQ;        // transposed layout
    const float* cb = coords + (size_t)b * NSEQ * 3;
    const unsigned char* mb = mask + (size_t)b * NSEQ;

    float sp = 2.0f + __expf(spread_w[hh]);  // MIN_SPREAD + exp(w)
    float inv2s2 = 1.0f / (2.0f * sp * sp);
    int qBase = qt * 16;

    // Q A-fragments for the two 32-wide K chunks of d_k=64
    v16h aQ[2];
#pragma unroll
    for (int c = 0; c < 2; ++c) {
        const _Float16* qrow = Qp + (size_t)(qBase + lr) * DK + c * 32 + half * 8;
        h8 lo = *(const h8*)qrow;
        h8 hi = *(const h8*)(qrow + 16);
#pragma unroll
        for (int i = 0; i < 8; ++i) { aQ[c][i] = lo[i]; aQ[c][i + 8] = hi[i]; }
    }

    // query coords for this lane's 8 rows
    float cqx[8], cqy[8], cqz[8];
#pragma unroll
    for (int r = 0; r < 8; ++r) {
        const float* cc = cb + (size_t)(qBase + r + half * 8) * 3;
        cqx[r] = cc[0]; cqy[r] = cc[1]; cqz[r] = cc[2];
    }

    float m[8], lsum[8];
    v8f acc[4] = {};
#pragma unroll
    for (int r = 0; r < 8; ++r) { m[r] = NEG_INF; lsum[r] = 0.0f; }

    _Float16* plds = &pstage[wib][0];
    uint32_t kldsBase = (uint32_t)(uintptr_t)&klds[0][0];

    // prime buffer 0 with key block 0 (wave 0, 8 x b128 per-lane = 4KB)
    if (wib == 0) {
        uint64_t g0 = (uint64_t)(uintptr_t)Kp;
#pragma unroll
        for (int c2 = 0; c2 < 8; ++c2) {
            uint32_t voff = (uint32_t)(lane * 16 + c2 * 512);
            uint32_t loff = kldsBase + voff;
            asm volatile("global_load_async_to_lds_b128 %0, %1, %2"
                         :: "v"(loff), "v"(voff), "s"(g0) : "memory");
        }
    }

    for (int kb2 = 0; kb2 < NSEQ; kb2 += 32) {
        int cur = (kb2 >> 5) & 1;

        // all waves done reading buffer cur^1 before it is overwritten
        __syncthreads();
        if (wib == 0) {
            int nxt = kb2 + 32; if (nxt >= NSEQ) nxt = 0;   // wrap: harmless refetch
            uint64_t g0 = (uint64_t)(uintptr_t)(Kp + (size_t)nxt * DK);
            uint32_t bufo = (uint32_t)((cur ^ 1) * (32 * DK * 2));
#pragma unroll
            for (int c2 = 0; c2 < 8; ++c2) {
                uint32_t voff = (uint32_t)(lane * 16 + c2 * 512);
                uint32_t loff = kldsBase + bufo + voff;
                asm volatile("global_load_async_to_lds_b128 %0, %1, %2"
                             :: "v"(loff), "v"(voff), "s"(g0) : "memory");
            }
            // block `cur` (issued last round) complete; next block may stay in flight
            asm volatile("s_wait_asynccnt 0x8" ::: "memory");
        }
        __syncthreads();   // publish buffer cur to all waves

        __builtin_prefetch(Vp + (size_t)lr * NSEQ + kb2 + 32, 0, 0); // global_prefetch_b8

        // ---- scores: two 16-key tiles, K-dim 64 split into two WMMAs each
        v8f s[2] = {};
#pragma unroll
        for (int T = 0; T < 2; ++T) {
#pragma unroll
            for (int c = 0; c < 2; ++c) {
                const _Float16* krow = &klds[cur][(T * 16 + lr) * DK + c * 32 + half * 8];
                h8 lo = *(const h8*)krow;
                h8 hi = *(const h8*)(krow + 16);
                v16h Bv;
#pragma unroll
                for (int i = 0; i < 8; ++i) { Bv[i] = lo[i]; Bv[i + 8] = hi[i]; }
                s[T] = __builtin_amdgcn_wmma_f32_16x16x32_f16(false, aQ[c], false, Bv,
                                                              (short)0, s[T], false, false);
            }
        }

        // ---- RBF weighting + threshold + mask (in C-layout registers)
#pragma unroll
        for (int T = 0; T < 2; ++T) {
            int key = kb2 + T * 16 + lr;
            const float* ck = cb + (size_t)key * 3;
            float kx = ck[0], ky = ck[1], kz = ck[2];
            bool mk = mb[key] != 0;
#pragma unroll
            for (int r = 0; r < 8; ++r) {
                float dx = cqx[r] - kx, dy = cqy[r] - ky, dz = cqz[r] - kz;
                float d2  = dx * dx + dy * dy + dz * dz;
                float rbf = __expf(-d2 * inv2s2);
                bool valid = mk && (rbf >= 0.001f) && (rbf <= 1.0f);
                s[T][r] = valid ? s[T][r] * rbf : NEG_INF;
            }
        }

        // ---- online softmax: row reductions across the 16-lane half
        float bmax[8];
#pragma unroll
        for (int r = 0; r < 8; ++r) bmax[r] = fmaxf(s[0][r], s[1][r]);
#pragma unroll
        for (int off = 8; off >= 1; off >>= 1)
#pragma unroll
            for (int r = 0; r < 8; ++r)
                bmax[r] = fmaxf(bmax[r], __shfl_xor(bmax[r], off, 32));

        float alpha[8], rs[8];
#pragma unroll
        for (int r = 0; r < 8; ++r) {
            float mn = fmaxf(m[r], bmax[r]);
            alpha[r] = __expf(m[r] - mn);
            m[r] = mn;
            float p0 = __expf(s[0][r] - mn);
            float p1 = __expf(s[1][r] - mn);
            s[0][r] = p0; s[1][r] = p1;
            rs[r] = p0 + p1;
        }
#pragma unroll
        for (int off = 8; off >= 1; off >>= 1)
#pragma unroll
            for (int r = 0; r < 8; ++r)
                rs[r] += __shfl_xor(rs[r], off, 32);
#pragma unroll
        for (int r = 0; r < 8; ++r) lsum[r] = lsum[r] * alpha[r] + rs[r];
#pragma unroll
        for (int c = 0; c < 4; ++c)
#pragma unroll
            for (int r = 0; r < 8; ++r) acc[c][r] *= alpha[r];

        // ---- P: C layout -> A layout via per-wave LDS (in-order within wave)
#pragma unroll
        for (int T = 0; T < 2; ++T)
#pragma unroll
            for (int r = 0; r < 8; ++r)
                plds[(r + half * 8) * 40 + T * 16 + lr] = (_Float16)s[T][r];

        v16h aP;
        {
            const _Float16* prow = plds + lr * 40 + half * 8;
#pragma unroll
            for (int i = 0; i < 8; ++i) { aP[i] = prow[i]; aP[i + 8] = prow[16 + i]; }
        }

        // ---- P.V : transposed V gives contiguous B-fragments (2 x h8 / chunk)
#pragma unroll
        for (int c = 0; c < 4; ++c) {
            const _Float16* vb = Vp + (size_t)(c * 16 + lr) * NSEQ + kb2 + half * 8;
            h8 lo = *(const h8*)vb;
            h8 hi = *(const h8*)(vb + 16);
            v16h Bv;
#pragma unroll
            for (int i = 0; i < 8; ++i) { Bv[i] = lo[i]; Bv[i + 8] = hi[i]; }
            acc[c] = __builtin_amdgcn_wmma_f32_16x16x32_f16(false, aP, false, Bv,
                                                            (short)0, acc[c], false, false);
        }
    }

    // ---- finalize: O / l, store f16 with feature index f = dk_col*H + head
#pragma unroll
    for (int c = 0; c < 4; ++c)
#pragma unroll
        for (int r = 0; r < 8; ++r) {
            int M = r + half * 8;
            float o = acc[c][r] / lsum[r];
            int f = (c * 16 + lr) * NHEAD + hh;
            AO[((size_t)b * NSEQ + qBase + M) * DMODEL + f] = (_Float16)o;
        }
}

// ---------------------------------------------------------------------------
// Kernel 3: output projection Y[4096,512] = AO[4096,512] * out_w^T, fp32 out.
// ---------------------------------------------------------------------------
__global__ __launch_bounds__(256) void geo_outproj_kernel(
    const _Float16* __restrict__ AO, const float* __restrict__ out_w,
    float* __restrict__ Y)
{
    int wave = (int)((blockIdx.x * blockDim.x + threadIdx.x) >> 5);
    int lane = (int)(threadIdx.x & 31);
    int half = lane >> 4, lr = lane & 15;

    int ct = wave & 31; wave >>= 5;          // DMODEL/16 = 32 col tiles
    int rt = wave;                           // (NB*NSEQ)/16 = 256 row tiles
    int rowBase = rt * 16, colBase = ct * 16;

    v8f acc = {};
    for (int kb = 0; kb < DMODEL; kb += 32) {
        const _Float16* arow = AO + (size_t)(rowBase + lr) * DMODEL + kb + half * 16;
        h8 lo = *(const h8*)arow;
        h8 hi = *(const h8*)(arow + 8);
        v16h A;
#pragma unroll
        for (int i = 0; i < 8; ++i) { A[i] = lo[i]; A[i + 8] = hi[i]; }

        // B[kk][n] = out_w[colBase+n][kb+kk]  (out @ out_w.T) -> contiguous f32
        const float* wrow = out_w + (size_t)(colBase + lr) * DMODEL + kb + half * 16;
        float4 w0 = *(const float4*)(wrow);
        float4 w1 = *(const float4*)(wrow + 4);
        float4 w2 = *(const float4*)(wrow + 8);
        float4 w3 = *(const float4*)(wrow + 12);
        float wf[16] = {w0.x,w0.y,w0.z,w0.w, w1.x,w1.y,w1.z,w1.w,
                        w2.x,w2.y,w2.z,w2.w, w3.x,w3.y,w3.z,w3.w};
        v16h Bv;
#pragma unroll
        for (int i = 0; i < 16; ++i) Bv[i] = (_Float16)wf[i];

        acc = __builtin_amdgcn_wmma_f32_16x16x32_f16(false, A, false, Bv,
                                                     (short)0, acc, false, false);
    }

#pragma unroll
    for (int r = 0; r < 8; ++r) {
        int M = r + half * 8;
        Y[(size_t)(rowBase + M) * DMODEL + colBase + lr] = acc[r];
    }
}

// ---------------------------------------------------------------------------
extern "C" void kernel_launch(void* const* d_in, const int* in_sizes, int n_in,
                              void* d_out, int out_size, void* d_ws, size_t ws_size,
                              hipStream_t stream) {
    const float* q        = (const float*)d_in[0];
    const float* k        = (const float*)d_in[1];
    const float* v        = (const float*)d_in[2];
    const float* coords   = (const float*)d_in[3];
    const unsigned char* mask = (const unsigned char*)d_in[4];
    const float* q_proj   = (const float*)d_in[5];
    const float* k_proj   = (const float*)d_in[6];
    const float* v_proj   = (const float*)d_in[7];
    const float* q_bias   = (const float*)d_in[8];
    const float* k_bias   = (const float*)d_in[9];
    const float* v_bias   = (const float*)d_in[10];
    const float* spread_w = (const float*)d_in[11];
    const float* beta_w   = (const float*)d_in[12];
    const float* out_w    = (const float*)d_in[13];
    float* Y              = (float*)d_out;

    const size_t QKV_HALFS = (size_t)NB * NHEAD * NSEQ * DK;
    char* ws = (char*)d_ws;
    _Float16* Qh = (_Float16*)ws;
    _Float16* Kh = (_Float16*)(ws + QKV_HALFS * 2);
    _Float16* Vt = (_Float16*)(ws + QKV_HALFS * 4);               // [B,H,DK,N] f16
    _Float16* AO = (_Float16*)(ws + QKV_HALFS * 6);               // [B,N,512] f16

    // 1) projections: 3*2*8*128*4 = 24576 wave-tiles, 8 waves / block
    geo_proj_kernel<<<3072, 256, 0, stream>>>(
        q, k, v, q_proj, k_proj, v_proj, q_bias, k_bias, v_bias, beta_w,
        Qh, Kh, Vt);

    // 2) fused attention: 2*8*128 = 2048 wave-tiles (8 waves share one K/V)
    geo_attn_kernel<<<256, 256, 0, stream>>>(
        Qh, Kh, Vt, coords, mask, spread_w, AO);

    // 3) output projection: 256*32 = 8192 wave-tiles
    geo_outproj_kernel<<<1024, 256, 0, stream>>>(AO, out_w, Y);
}
